// YOGO_1958505087274
// MI455X (gfx1250) — matrix operand order repaired and checked by reading, not compiled, not run on hardware
//
#include <hip/hip_runtime.h>

// ---------------- model constants (match reference) ----------------
#define NB    4
#define NPTS  16384
#define INC   22
#define TL    128   // TOKEN_L
#define TS    32    // TOKEN_S
#define TC    256   // TOKEN_C
#define NHEAD 2
#define NCLS  50

// ---------------- WMMA GEMM tile config ----------------
#define GTM 128     // block tile M
#define GTN 64      // block tile N
#define GTK 32      // k step
#define LDT 40      // padded LDS row stride (half-words): 16B-aligned rows, conflict-free frags

typedef __attribute__((ext_vector_type(16))) __bf16 v16bf;
typedef __attribute__((ext_vector_type(8)))  float  v8f;

union Frag { v16bf v; uint4 q[2]; };

__device__ __forceinline__ unsigned short f2bf(float f) {
  unsigned int u = __float_as_uint(f);
  u += 0x7FFFu + ((u >> 16) & 1u);     // round-to-nearest-even
  return (unsigned short)(u >> 16);
}

// Batched/strided GEMM: D[b] = op(A[b]) * op(B[b])  (+ ReLU) (+ residual R[b])
// op(A): logical [M x K]; stored row-major [M x K] (lda) or, if TA, [K x M] (lda).
// op(B): logical [K x N]; stored row-major [K x N] (ldb) or, if TB, [N x K] (ldb).
// D,R: row-major [M x N] (ld = N). flags: 1 = relu(result), 2 = add R.
template <int TA, int TB>
__global__ __launch_bounds__(256) void gemm_bf16_wmma(
    const float* __restrict__ A, const float* __restrict__ Bm,
    const float* __restrict__ R, float* __restrict__ D,
    int M, int N, int K,
    long long sAb, long long sBb, long long sRb, long long sDb,
    int lda, int ldb, int flags)
{
  __shared__ __attribute__((aligned(16))) unsigned short sA[GTM * LDT];
  __shared__ __attribute__((aligned(16))) unsigned short sBt[GTN * LDT]; // [n][k]

  const int bz = blockIdx.z;
  A  += (long long)bz * sAb;
  Bm += (long long)bz * sBb;
  if (R) R += (long long)bz * sRb;
  D  += (long long)bz * sDb;

  const int m0   = blockIdx.y * GTM;
  const int n0   = blockIdx.x * GTN;
  const int tid  = threadIdx.x;
  const int wid  = tid >> 5;
  const int lane = tid & 31;
  const int half = lane >> 4;
  const int rm   = lane & 15;
  const int mw   = (wid & 3) * 16;     // wave M sub-tiles: mw, mw+64
  const int nw   = (wid >> 2) * 16;    // wave N sub-tiles: nw, nw+32

  const bool mFull = (m0 + GTM <= M);
  const bool nFull = (n0 + GTN <= N);

  v8f acc00 = {}, acc01 = {}, acc10 = {}, acc11 = {};

  for (int k0 = 0; k0 < K; k0 += GTK) {
    const bool kFull = (k0 + GTK <= K);

    // ---------------- stage A tile (128x32) -> sA[m][k] ----------------
    if (mFull && kFull && ((lda & 3) == 0)) {
      if (!TA) {
        #pragma unroll
        for (int i = 0; i < (GTM * GTK) / (256 * 4); ++i) {   // 4 iters, float4 coalesced
          int lin = (tid + i * 256) * 4;
          int am = lin >> 5, ak = lin & 31;
          float4 v = *(const float4*)(A + (long long)(m0 + am) * lda + (k0 + ak));
          ushort4 h; h.x = f2bf(v.x); h.y = f2bf(v.y); h.z = f2bf(v.z); h.w = f2bf(v.w);
          *(ushort4*)&sA[am * LDT + ak] = h;
        }
      } else {
        #pragma unroll
        for (int i = 0; i < (GTM * GTK) / (256 * 4); ++i) {
          int lin = (tid + i * 256) * 4;
          int ak = lin >> 7, am = lin & 127;
          float4 v = *(const float4*)(A + (long long)(k0 + ak) * lda + (m0 + am));
          sA[(am + 0) * LDT + ak] = f2bf(v.x);
          sA[(am + 1) * LDT + ak] = f2bf(v.y);
          sA[(am + 2) * LDT + ak] = f2bf(v.z);
          sA[(am + 3) * LDT + ak] = f2bf(v.w);
        }
      }
    } else {
      // branchless edge path: clamp OOB index to 0, unconditional load, select 0
      #pragma unroll 1
      for (int i = 0; i < (GTM * GTK) / 256; ++i) {
        int idx = tid + i * 256;
        int am, ak;
        if (!TA) { am = idx >> 5;  ak = idx & 31; }
        else     { am = idx & 127; ak = idx >> 7; }
        int gm = m0 + am, gk = k0 + ak;
        bool ok = (gm < M) && (gk < K);
        int cm = ok ? gm : 0;
        int ck = ok ? gk : 0;
        float v = TA ? A[(long long)ck * lda + cm] : A[(long long)cm * lda + ck];
        v = ok ? v : 0.f;
        sA[am * LDT + ak] = f2bf(v);
      }
    }

    // ---------------- stage B tile (32x64) transposed -> sBt[n][k] ----------------
    if (nFull && kFull && ((ldb & 3) == 0)) {
      if (!TB) {
        #pragma unroll
        for (int i = 0; i < (GTN * GTK) / (256 * 4); ++i) {   // 2 iters, float4 coalesced
          int lin = (tid + i * 256) * 4;
          int bk = lin >> 6, bn = lin & 63;
          float4 v = *(const float4*)(Bm + (long long)(k0 + bk) * ldb + (n0 + bn));
          sBt[(bn + 0) * LDT + bk] = f2bf(v.x);
          sBt[(bn + 1) * LDT + bk] = f2bf(v.y);
          sBt[(bn + 2) * LDT + bk] = f2bf(v.z);
          sBt[(bn + 3) * LDT + bk] = f2bf(v.w);
        }
      } else {
        #pragma unroll
        for (int i = 0; i < (GTN * GTK) / (256 * 4); ++i) {
          int lin = (tid + i * 256) * 4;
          int bn = lin >> 5, bk = lin & 31;
          float4 v = *(const float4*)(Bm + (long long)(n0 + bn) * ldb + (k0 + bk));
          ushort4 h; h.x = f2bf(v.x); h.y = f2bf(v.y); h.z = f2bf(v.z); h.w = f2bf(v.w);
          *(ushort4*)&sBt[bn * LDT + bk] = h;
        }
      }
    } else {
      #pragma unroll 1
      for (int i = 0; i < (GTN * GTK) / 256; ++i) {
        int idx = tid + i * 256;
        int bn, bk;
        if (!TB) { bn = idx & 63; bk = idx >> 6; }
        else     { bk = idx & 31; bn = idx >> 5; }
        int gn = n0 + bn, gk = k0 + bk;
        bool ok = (gn < N) && (gk < K);
        int cn = ok ? gn : 0;
        int ck = ok ? gk : 0;
        float v = TB ? Bm[(long long)cn * ldb + ck] : Bm[(long long)ck * ldb + cn];
        v = ok ? v : 0.f;
        sBt[bn * LDT + bk] = f2bf(v);
      }
    }
    __syncthreads();

    // A fragment: lane (half,rm) row M, K = {half*8+0..7, 16+half*8+0..7}
    Frag fa0, fa1, fb0, fb1;
    const unsigned short* pa0 = &sA[(mw + rm) * LDT];
    fa0.q[0] = *(const uint4*)(pa0 + half * 8);
    fa0.q[1] = *(const uint4*)(pa0 + 16 + half * 8);
    const unsigned short* pa1 = &sA[(mw + 64 + rm) * LDT];
    fa1.q[0] = *(const uint4*)(pa1 + half * 8);
    fa1.q[1] = *(const uint4*)(pa1 + 16 + half * 8);
    // B fragment: lane col N, K = {half*8+0..7, 16+half*8+0..7}
    const unsigned short* pb0 = &sBt[(nw + rm) * LDT + half * 8];
    fb0.q[0] = *(const uint4*)(pb0);
    fb0.q[1] = *(const uint4*)(pb0 + 16);
    const unsigned short* pb1 = &sBt[(nw + 32 + rm) * LDT + half * 8];
    fb1.q[0] = *(const uint4*)(pb1);
    fb1.q[1] = *(const uint4*)(pb1 + 16);

    acc00 = __builtin_amdgcn_wmma_f32_16x16x32_bf16(false, fa0.v, false, fb0.v,
                                                    (short)0, acc00, false, false);
    acc01 = __builtin_amdgcn_wmma_f32_16x16x32_bf16(false, fa0.v, false, fb1.v,
                                                    (short)0, acc01, false, false);
    acc10 = __builtin_amdgcn_wmma_f32_16x16x32_bf16(false, fa1.v, false, fb0.v,
                                                    (short)0, acc10, false, false);
    acc11 = __builtin_amdgcn_wmma_f32_16x16x32_bf16(false, fa1.v, false, fb1.v,
                                                    (short)0, acc11, false, false);
    __syncthreads();
  }

  // C/D layout: VGPR r, lane -> M = r + half*8, N = rm
  #pragma unroll
  for (int r = 0; r < 8; ++r) {
    int gmA = m0 + mw + r + half * 8;
    int gmB = gmA + 64;
    int gn0 = n0 + nw + rm;
    int gn1 = gn0 + 32;
    if (gmA < M) {
      if (gn0 < N) {
        float v = acc00[r];
        if (flags & 1) v = fmaxf(v, 0.f);
        if (flags & 2) v += R[(long long)gmA * N + gn0];
        D[(long long)gmA * N + gn0] = v;
      }
      if (gn1 < N) {
        float v = acc01[r];
        if (flags & 1) v = fmaxf(v, 0.f);
        if (flags & 2) v += R[(long long)gmA * N + gn1];
        D[(long long)gmA * N + gn1] = v;
      }
    }
    if (gmB < M) {
      if (gn0 < N) {
        float v = acc10[r];
        if (flags & 1) v = fmaxf(v, 0.f);
        if (flags & 2) v += R[(long long)gmB * N + gn0];
        D[(long long)gmB * N + gn0] = v;
      }
      if (gn1 < N) {
        float v = acc11[r];
        if (flags & 1) v = fmaxf(v, 0.f);
        if (flags & 2) v += R[(long long)gmB * N + gn1];
        D[(long long)gmB * N + gn1] = v;
      }
    }
  }
}

// softmax over middle axis of (outer, n, inner); element i of column (o,j) at
// d[o*n*inner + i*inner + j]; logits pre-scaled by `scale`.
__global__ __launch_bounds__(128) void softmax_mid(float* __restrict__ d,
                                                   int n, int inner, float scale)
{
  long long bid = blockIdx.x;
  long long o = bid / inner;
  long long j = bid % inner;
  float* base = d + o * (long long)n * inner + j;
  __shared__ float red[128];
  int tid = threadIdx.x;

  float mx = -3.4e38f;
  for (int i = tid; i < n; i += 128) mx = fmaxf(mx, base[(long long)i * inner] * scale);
  red[tid] = mx; __syncthreads();
  for (int s = 64; s > 0; s >>= 1) { if (tid < s) red[tid] = fmaxf(red[tid], red[tid + s]); __syncthreads(); }
  mx = red[0]; __syncthreads();

  float sum = 0.f;
  for (int i = tid; i < n; i += 128) {
    float e = __expf(base[(long long)i * inner] * scale - mx);
    base[(long long)i * inner] = e;
    sum += e;
  }
  red[tid] = sum; __syncthreads();
  for (int s = 64; s > 0; s >>= 1) { if (tid < s) red[tid] += red[tid + s]; __syncthreads(); }
  float inv = 1.f / red[0];
  for (int i = tid; i < n; i += 128) base[(long long)i * inner] *= inv;
}

// Farthest-point sampling: one block per batch; mind is (B,N) global scratch.
__global__ __launch_bounds__(256) void fps_kernel(const float* __restrict__ x,
                                                  float* __restrict__ mind,
                                                  int* __restrict__ cidx,
                                                  int cin, int n, int l)
{
  int b = blockIdx.x;
  const float* px = x + (long long)b * cin * n;
  const float* py = px + n;
  const float* pz = px + 2 * n;
  float* md = mind + (long long)b * n;
  int tid = threadIdx.x;
  __shared__ float rv[256];
  __shared__ int   ri[256];
  __shared__ float sc[3];

  float x0 = px[0], y0 = py[0], z0 = pz[0];
  for (int i = tid; i < n; i += 256) {
    float dx = px[i] - x0, dy = py[i] - y0, dz = pz[i] - z0;
    md[i] = dx * dx + dy * dy + dz * dz;
  }
  if (tid == 0) cidx[(long long)b * l] = 0;
  __syncthreads();

  for (int it = 1; it < l; ++it) {
    float bv = -1.f; int bi = 0;
    for (int i = tid; i < n; i += 256) {
      float v = md[i];
      if (v > bv) { bv = v; bi = i; }
    }
    rv[tid] = bv; ri[tid] = bi; __syncthreads();
    for (int s = 128; s > 0; s >>= 1) {
      if (tid < s) {
        if (rv[tid + s] > rv[tid] || (rv[tid + s] == rv[tid] && ri[tid + s] < ri[tid])) {
          rv[tid] = rv[tid + s]; ri[tid] = ri[tid + s];
        }
      }
      __syncthreads();
    }
    if (tid == 0) {
      int nx = ri[0];
      cidx[(long long)b * l + it] = nx;
      sc[0] = px[nx]; sc[1] = py[nx]; sc[2] = pz[nx];
    }
    __syncthreads();
    float cx = sc[0], cy = sc[1], cz = sc[2];
    for (int i = tid; i < n; i += 256) {
      float dx = px[i] - cx, dy = py[i] - cy, dz = pz[i] - cz;
      float dd = dx * dx + dy * dy + dz * dz;
      md[i] = fminf(md[i], dd);
    }
    __syncthreads();
  }
}

// kNN: one block per (b,l) center; dist is (B*L, N) global scratch; 32 argmin sweeps.
__global__ __launch_bounds__(256) void knn_kernel(const float* __restrict__ x,
                                                  const int* __restrict__ cidx,
                                                  float* __restrict__ dist,
                                                  int* __restrict__ knn,
                                                  int cin, int n, int l, int kk)
{
  int bl = blockIdx.x; int b = bl / l;
  const float* px = x + (long long)b * cin * n;
  const float* py = px + n;
  const float* pz = px + 2 * n;
  float* dd = dist + (long long)bl * n;
  int tid = threadIdx.x;
  __shared__ float rv[256];
  __shared__ int   ri[256];

  int ci = cidx[bl];
  float cx = px[ci], cy = py[ci], cz = pz[ci];
  for (int i = tid; i < n; i += 256) {
    float dx = px[i] - cx, dy = py[i] - cy, dz = pz[i] - cz;
    dd[i] = dx * dx + dy * dy + dz * dz;
  }
  __syncthreads();

  for (int s = 0; s < kk; ++s) {
    float bv = 3.4e38f; int bi = 0;
    for (int i = tid; i < n; i += 256) {
      float v = dd[i];
      if (v < bv) { bv = v; bi = i; }
    }
    rv[tid] = bv; ri[tid] = bi; __syncthreads();
    for (int t = 128; t > 0; t >>= 1) {
      if (tid < t) {
        if (rv[tid + t] < rv[tid] || (rv[tid + t] == rv[tid] && ri[tid + t] < ri[tid])) {
          rv[tid] = rv[tid + t]; ri[tid] = ri[tid + t];
        }
      }
      __syncthreads();
    }
    if (tid == 0) { knn[(long long)bl * kk + s] = ri[0]; dd[ri[0]] = 3.4e38f; }
    __syncthreads();
  }
}

// out[b][c][l] = max_k f[b][c][knn[b][l][k]]
__global__ void gather_max_kernel(const float* __restrict__ f, const int* __restrict__ knn,
                                  float* __restrict__ out, int C, int n, int l, int kk)
{
  int bl = blockIdx.y; int b = bl / l; int li = bl % l;
  int c = blockIdx.x * blockDim.x + threadIdx.x;
  if (c >= C) return;
  const int* kn = knn + (long long)bl * kk;
  const float* fr = f + ((long long)b * C + c) * n;
  float m = -3.4e38f;
  for (int k = 0; k < kk; ++k) m = fmaxf(m, fr[kn[k]]);
  out[((long long)b * C + c) * l + li] = m;
}

// ---------------- host side ----------------
struct RimW {
  const float *df1, *df2, *tk, *tq, *tv, *ff1, *ff2, *pv, *pk, *pq, *fb;
  int ci, co;
};

static void gemm(hipStream_t st, const float* A, const float* B, const float* R, float* D,
                 int M, int N, int K, int batch,
                 long long sAb, long long sBb, long long sRb, long long sDb,
                 int lda, int ldb, int tA, int tB, int flags)
{
  dim3 g((unsigned)((N + GTN - 1) / GTN), (unsigned)((M + GTM - 1) / GTM), (unsigned)batch);
  if (tA)
    gemm_bf16_wmma<1, 0><<<g, dim3(256), 0, st>>>(A, B, R, D, M, N, K, sAb, sBb, sRb, sDb, lda, ldb, flags);
  else if (tB)
    gemm_bf16_wmma<0, 1><<<g, dim3(256), 0, st>>>(A, B, R, D, M, N, K, sAb, sBb, sRb, sDb, lda, ldb, flags);
  else
    gemm_bf16_wmma<0, 0><<<g, dim3(256), 0, st>>>(A, B, R, D, M, N, K, sAb, sBb, sRb, sDb, lda, ldb, flags);
}

static void softmax(hipStream_t st, float* d, long long outer, int n, int inner, float scale)
{
  dim3 g((unsigned)(outer * inner));
  softmax_mid<<<g, dim3(128), 0, st>>>(d, n, inner, scale);
}

extern "C" void kernel_launch(void* const* d_in, const int* in_sizes, int n_in,
                              void* d_out, int out_size, void* d_ws, size_t ws_size,
                              hipStream_t stream)
{
  (void)out_size; (void)ws_size;
  const int cs[5] = {32, 64, 128, 256, 512};

  // ---- resolve input pointer table (handles both flatten orders) ----
  RimW rims[8];
  for (int s = 0; s < 4; ++s) {
    rims[2 * s].ci = cs[s];     rims[2 * s].co = cs[s];
    rims[2 * s + 1].ci = cs[s]; rims[2 * s + 1].co = cs[s + 1];
  }
  const float *cls1 = nullptr, *cls2 = nullptr, *stem1 = nullptr, *stem2 = nullptr, *xin = nullptr;
  const long long xElems = (long long)NB * INC * NPTS;
  bool xLast = (n_in > 0) && ((long long)in_sizes[n_in - 1] == xElems);
  auto F = [&](int i) { return (const float*)d_in[i]; };
  if (xLast) {
    // jax pytree order: sorted dict keys, 'params' before 'x'
    int i = 0;
    cls1 = F(i++); cls2 = F(i++);
    for (int r = 0; r < 8; ++r) {
      RimW& m = rims[r];
      bool hasFb = (r & 1);
      m.df1 = F(i++); m.df2 = F(i++);
      m.fb = hasFb ? F(i++) : nullptr;
      m.ff1 = F(i++); m.ff2 = F(i++);
      m.pk = F(i++); m.pq = F(i++); m.pv = F(i++);
      m.tk = F(i++); m.tq = F(i++); m.tv = F(i++);
    }
    stem1 = F(i++); stem2 = F(i++); xin = F(i++);
  } else {
    // python insertion order: x first, then params in construction order
    int i = 0;
    xin = F(i++); stem1 = F(i++); stem2 = F(i++); cls1 = F(i++); cls2 = F(i++);
    for (int r = 0; r < 8; ++r) {
      RimW& m = rims[r];
      bool hasFb = (r & 1);
      m.df1 = F(i++); m.df2 = F(i++);
      m.tk = F(i++); m.tq = F(i++); m.tv = F(i++);
      m.ff1 = F(i++); m.ff2 = F(i++);
      m.pv = F(i++); m.pk = F(i++); m.pq = F(i++);
      m.fb = hasFb ? F(i++) : nullptr;
    }
  }

  // ---- workspace carve (bump allocator, 256B aligned) ----
  char* wp = (char*)d_ws;
  size_t off = 0;
  auto alloc = [&](size_t bytes) -> void* {
    off = (off + 255) & ~(size_t)255;
    void* p = wp + off;
    off += bytes;
    return p;
  };
  float* G0    = (float*)alloc((size_t)NB * 512 * NPTS * 4);
  float* G1    = (float*)alloc((size_t)NB * 512 * NPTS * 4);
  float* coef  = (float*)alloc((size_t)NB * NHEAD * NPTS * TL * 4);
  float* tok0  = (float*)alloc((size_t)NB * TC * TL * 4);
  float* tok1  = (float*)alloc((size_t)NB * TC * TL * 4);
  float* pool  = (float*)alloc((size_t)NB * TC * TL * 4);
  float* t1b   = (float*)alloc((size_t)NB * TC * TL * 4);
  float* kb    = (float*)alloc((size_t)NB * 128 * TL * 4);
  float* qb    = (float*)alloc((size_t)NB * 128 * TL * 4);
  float* vb    = (float*)alloc((size_t)NB * 256 * TL * 4);
  float* kqb   = (float*)alloc((size_t)NB * NHEAD * TL * TL * 4);
  float* xat   = (float*)alloc((size_t)NB * 256 * TL * 4);
  float* h1b   = (float*)alloc((size_t)NB * 512 * TL * 4);
  float* pvt   = (float*)alloc((size_t)NB * 512 * TL * 4);
  float* pkt   = (float*)alloc((size_t)NB * 512 * TL * 4);
  int*   cidx  = (int*)alloc((size_t)NB * TL * 4);
  int*   knnI  = (int*)alloc((size_t)NB * TL * TS * 4);
  float* fmind = (float*)alloc((size_t)NB * NPTS * 4);
  float* kdist = (float*)alloc((size_t)NB * TL * NPTS * 4);

  // ---- FPS + kNN (coords = first 3 channels of x) ----
  fps_kernel<<<dim3(NB), dim3(256), 0, stream>>>(xin, fmind, cidx, INC, NPTS, TL);
  knn_kernel<<<dim3(NB * TL), dim3(256), 0, stream>>>(xin, cidx, kdist, knnI, INC, NPTS, TL, TS);

  // ---- stem: f = stem2 * relu(stem1 * x) ----
  gemm(stream, stem1, xin, nullptr, G1, 32, NPTS, INC, NB,
       0, (long long)INC * NPTS, 0, 32LL * NPTS, INC, NPTS, 0, 0, 1);
  gemm(stream, stem2, G1, nullptr, G0, 32, NPTS, 32, NB,
       0, 32LL * NPTS, 0, 32LL * NPTS, 32, NPTS, 0, 0, 0);

  float* f = G0;
  float* other = G1;
  const float* tokPrev = nullptr;

  for (int r = 0; r < 8; ++r) {
    const RimW& w = rims[r];
    const int ci = w.ci, co = w.co, ch = co / NHEAD;

    // 1) pooled = max_k gather(f, knn)           (B, ci, TL)
    gather_max_kernel<<<dim3((ci + 127) / 128, NB * TL), dim3(128), 0, stream>>>(
        f, knnI, pool, ci, NPTS, TL, TS);

    // 2) t1 = relu(df1 * pooled)                 (B, TC, TL)
    gemm(stream, w.df1, pool, nullptr, t1b, TC, TL, ci, NB,
         0, (long long)ci * TL, 0, (long long)TC * TL, ci, TL, 0, 0, 1);

    // 3) tokens = df2 * t1 (+ tokens_prev)
    float* tok = (tokPrev == tok0) ? tok1 : tok0;
    gemm(stream, w.df2, t1b, tokPrev, tok, TC, TL, TC, NB,
         0, (long long)TC * TL, (long long)TC * TL, (long long)TC * TL,
         TC, TL, 0, 0, tokPrev ? 2 : 0);

    // 4) transformer on tokens
    gemm(stream, w.tk, tok, nullptr, kb, 128, TL, TC, NB,
         0, (long long)TC * TL, 0, 128LL * TL, TC, TL, 0, 0, 0);
    gemm(stream, w.tq, tok, nullptr, qb, 128, TL, TC, NB,
         0, (long long)TC * TL, 0, 128LL * TL, TC, TL, 0, 0, 0);
    gemm(stream, w.tv, tok, nullptr, vb, 256, TL, TC, NB,
         0, (long long)TC * TL, 0, 256LL * TL, TC, TL, 0, 0, 0);
    // kq[l][m] = K^T Q per (b,h): K,Q stored c-major -> transA
    gemm(stream, kb, qb, nullptr, kqb, TL, TL, 64, NB * NHEAD,
         64LL * TL, 64LL * TL, 0, (long long)TL * TL, TL, TL, 1, 0, 0);
    softmax(stream, kqb, NB * NHEAD, TL, TL, 1.f / sqrtf((float)TL));  // axis = l
    // xattn = V * KQ + tokens (per half-channel bh slice)
    gemm(stream, vb, kqb, tok, xat, 128, TL, TL, NB * NHEAD,
         128LL * TL, (long long)TL * TL, 128LL * TL, 128LL * TL, TL, TL, 0, 0, 2);
    // ff: tokens = xattn + ff2 * relu(ff1 * xattn)
    gemm(stream, w.ff1, xat, nullptr, h1b, 2 * TC, TL, TC, NB,
         0, (long long)TC * TL, 0, 2LL * TC * TL, TC, TL, 0, 0, 1);
    gemm(stream, w.ff2, h1b, xat, tok, TC, TL, 2 * TC, NB,
         0, 2LL * TC * TL, (long long)TC * TL, (long long)TC * TL, 2 * TC, TL, 0, 0, 2);

    // 5) projector
    gemm(stream, w.pv, tok, nullptr, pvt, co, TL, TC, NB,
         0, (long long)TC * TL, 0, (long long)co * TL, TC, TL, 0, 0, 0);
    gemm(stream, w.pk, tok, nullptr, pkt, co, TL, TC, NB,
         0, (long long)TC * TL, 0, (long long)co * TL, TC, TL, 0, 0, 0);

    const float* f2; float* qbig; float* fdst;
    if (w.fb) {
      gemm(stream, w.fb, f, nullptr, other, co, NPTS, ci, NB,
           0, (long long)ci * NPTS, 0, (long long)co * NPTS, ci, NPTS, 0, 0, 0);
      f2 = other; qbig = f; fdst = f;       // f dead after fb; q dead after logits
    } else {
      f2 = f; qbig = other; fdst = other;
    }
    // q = pq * f2                                (B, co, N)
    gemm(stream, w.pq, f2, nullptr, qbig, co, NPTS, co, NB,
         0, (long long)co * NPTS, 0, (long long)co * NPTS, co, NPTS, 0, 0, 0);
    // logits[n][l] = q^T k per (b,h)             (BH, N, TL)
    gemm(stream, qbig, pkt, nullptr, coef, NPTS, TL, ch, NB * NHEAD,
         (long long)ch * NPTS, (long long)ch * TL, 0, (long long)NPTS * TL,
         NPTS, TL, 1, 0, 0);
    softmax(stream, coef, (long long)NB * NHEAD * NPTS, TL, 1, 1.f / sqrtf((float)ch));
    // f_out = V * coef^T + f2                    (B, co, N)
    gemm(stream, pvt, coef, f2, fdst, ch, NPTS, TL, NB * NHEAD,
         (long long)ch * TL, (long long)NPTS * TL, (long long)ch * NPTS,
         (long long)ch * NPTS, TL, TL, 0, 1, 2);

    if (!w.fb) { float* t = f; f = other; other = t; }  // fb case: f stays, other dead
    tokPrev = tok;
  }

  // ---- classifier: out = cls2 * relu(cls1 * f) ----
  gemm(stream, cls1, f, nullptr, other, 512, NPTS, 512, NB,
       0, 512LL * NPTS, 0, 512LL * NPTS, 512, NPTS, 0, 0, 1);
  gemm(stream, cls2, other, nullptr, (float*)d_out, NCLS, NPTS, 512, NB,
       0, 512LL * NPTS, 0, (long long)NCLS * NPTS, 512, NPTS, 0, 0, 0);
}
